// LSTMPrediction_3728031613437
// MI455X (gfx1250) — compile-verified, matrix operand order checked by reading, not run on hardware
//
#include <hip/hip_runtime.h>
#include <hip/hip_bf16.h>

typedef __attribute__((ext_vector_type(16))) _Float16 v16h;
typedef __attribute__((ext_vector_type(8)))  _Float16 v8h;
typedef __attribute__((ext_vector_type(8)))  float    v8f;

#define STEPS 96
#define BATCH 2048
#define F_IN  96
#define HID   256
#define GATES 1024          // 4*HID
#define K0    352           // F_IN + HID  (layer0 concat K)
#define K1    512           // HID + HID   (layer1 concat K)
#define KB0   11            // K0/32
#define KB1   16            // K1/32
#define MT    16            // batch rows per workgroup
#define NTW   2             // n-tiles per gate per wave (8 waves * 2 * 16 = 256 cols/gate)

#define PACK0_HALFS (64 * KB0 * 32 * 16)   // 360448
#define PACK1_HALFS (64 * KB1 * 32 * 16)   // 524288

// Branchless, hardware-native activations: v_exp_f32 + v_rcp_f32 only.
static __device__ __forceinline__ float fast_sig(float x) {
  return __builtin_amdgcn_rcpf(1.0f + __expf(-x));
}
static __device__ __forceinline__ float fast_tanh(float x) {
  // tanh(x) = 2*sigmoid(2x) - 1
  return __builtin_amdgcn_rcpf(1.0f + __expf(-2.0f * x)) * 2.0f - 1.0f;
}

// ---------------------------------------------------------------------------
// Prep: convert f32 weights to f16 and pre-swizzle into WMMA B-fragment order.
// Fragment layout per 32x16 (KxN) tile: lane l (l15=N, hi=l>>4), 16 halfs:
//   e<8  -> K = 32*kb + hi*8 + e
//   e>=8 -> K = 32*kb + hi*8 + 16 + (e-8)
// pack index = ((nt*KB + kb)*32 + lane)*16 + e
// ---------------------------------------------------------------------------
__global__ void pack_weights_kernel(const float* __restrict__ Wih0,
                                    const float* __restrict__ Whh0,
                                    const float* __restrict__ Wih1,
                                    const float* __restrict__ Whh1,
                                    _Float16* __restrict__ pack0,
                                    _Float16* __restrict__ pack1) {
  int t = blockIdx.x * blockDim.x + threadIdx.x;
  if (t < PACK0_HALFS) {
    int e    = t & 15;
    int lane = (t >> 4) & 31;
    int tk   = t >> 9;            // nt*KB0 + kb
    int kb   = tk % KB0;
    int nt   = tk / KB0;
    int l15  = lane & 15;
    int hi   = lane >> 4;
    int k32  = hi * 8 + (e & 7) + ((e >> 3) << 4);
    int n    = nt * 16 + l15;
    int k    = kb * 32 + k32;
    float v  = (k < F_IN) ? Wih0[n * F_IN + k] : Whh0[n * HID + (k - F_IN)];
    pack0[t] = (_Float16)v;
  } else if (t < PACK0_HALFS + PACK1_HALFS) {
    int u    = t - PACK0_HALFS;
    int e    = u & 15;
    int lane = (u >> 4) & 31;
    int tk   = u >> 9;            // nt*KB1 + kb
    int kb   = tk & (KB1 - 1);
    int nt   = tk >> 4;
    int l15  = lane & 15;
    int hi   = lane >> 4;
    int k32  = hi * 8 + (e & 7) + ((e >> 3) << 4);
    int n    = nt * 16 + l15;
    int k    = kb * 32 + k32;
    float v  = (k < HID) ? Wih1[n * HID + k] : Whh1[n * HID + (k - HID)];
    pack1[u] = (_Float16)v;
  }
}

// ---------------------------------------------------------------------------
// Persistent LSTM rollout: one workgroup = 16 batch rows for all 96 steps.
// 8 waves; wave w owns gate-matched n-tiles {g*16 + 2w, g*16 + 2w + 1}, g=0..3,
// so i/f/g/o pre-activations for each output element sit in the same lane and
// the whole gate nonlinearity runs in registers (c-state lives in VGPRs).
// ---------------------------------------------------------------------------
__launch_bounds__(256, 1)
__global__ void lstm_persistent_kernel(const _Float16* __restrict__ pack0,
                                       const _Float16* __restrict__ pack1,
                                       const float* __restrict__ x,
                                       const float* __restrict__ b_ih0,
                                       const float* __restrict__ b_hh0,
                                       const float* __restrict__ b_ih1,
                                       const float* __restrict__ b_hh1,
                                       const float* __restrict__ fc_w,
                                       const float* __restrict__ fc_b,
                                       float* __restrict__ out) {
  __shared__ __attribute__((aligned(16))) _Float16 sXH0[MT][K0];  // [xw | h0]
  __shared__ __attribute__((aligned(16))) _Float16 sXH1[MT][K1];  // [h0 | h1]
  __shared__ float sB0[GATES];
  __shared__ float sB1[GATES];
  __shared__ float sFC[HID];
  __shared__ float sRed[MT][16];   // fc partial-dot reduction

  const int tid  = threadIdx.x;
  const int lane = tid & 31;
  const int w    = tid >> 5;      // wave 0..7
  const int l15  = lane & 15;
  const int hi   = lane >> 4;
  const int rowBase = blockIdx.x * MT;
  const int fr   = tid >> 4;      // fc phase: row 0..15
  const int fi   = tid & 15;      // fc phase: 16 threads per row

  // ---- init: xw window from x, h=0, fused biases, fc weights ----
  for (int idx = tid; idx < MT * K0; idx += 256) {
    int r = idx / K0, j = idx - r * K0;
    sXH0[r][j] = (j < F_IN) ? (_Float16)x[(rowBase + r) * F_IN + j]
                            : (_Float16)0.0f;
  }
  for (int idx = tid; idx < MT * K1; idx += 256) {
    int r = idx / K1, j = idx - r * K1;
    sXH1[r][j] = (_Float16)0.0f;
  }
  for (int k = tid; k < GATES; k += 256) {
    sB0[k] = b_ih0[k] + b_hh0[k];
    sB1[k] = b_ih1[k] + b_hh1[k];
  }
  for (int k = tid; k < HID; k += 256) sFC[k] = fc_w[k];

  float c0v[NTW][8];
  float c1v[NTW][8];
#pragma unroll
  for (int j = 0; j < NTW; ++j)
#pragma unroll
    for (int r = 0; r < 8; ++r) { c0v[j][r] = 0.0f; c1v[j][r] = 0.0f; }

  const float fcb = fc_b[0];
  __syncthreads();

#pragma unroll 1
  for (int step = 0; step < STEPS; ++step) {
    // ================= layer 0 GEMM: z0 = [xw|h0] @ W0cat^T =================
    v8f acc0[4][NTW] = {};
    {
      const _Float16* arow = &sXH0[l15][0];
#pragma unroll 1
      for (int kb = 0; kb < KB0; ++kb) {
        union { v16h v; v8h h[2]; } a;
        a.h[0] = *(const v8h*)(arow + kb * 32 + hi * 8);
        a.h[1] = *(const v8h*)(arow + kb * 32 + hi * 8 + 16);
#pragma unroll
        for (int g = 0; g < 4; ++g) {
#pragma unroll
          for (int j = 0; j < NTW; ++j) {
            const int nt = g * 16 + (w << 1) + j;
            const _Float16* bp = pack0 + ((((size_t)nt * KB0 + kb) * 32 + lane) << 4);
            union { v16h v; v8h h[2]; } b;
            b.h[0] = *(const v8h*)bp;
            b.h[1] = *(const v8h*)(bp + 8);
            acc0[g][j] = __builtin_amdgcn_wmma_f32_16x16x32_f16(
                false, a.v, false, b.v, (short)0, acc0[g][j], false, false);
          }
        }
      }
    }
    __syncthreads();   // all waves done reading sXH0 before h0 overwrites

    // ================= layer 0 gates (branchless, registers only) ===========
#pragma unroll
    for (int j = 0; j < NTW; ++j) {
      const int colb = ((w << 1) + j) * 16 + l15;   // n in [0, 256)
      const float bi = sB0[colb];
      const float bf = sB0[256 + colb];
      const float bg = sB0[512 + colb];
      const float bo = sB0[768 + colb];
#pragma unroll
      for (int r = 0; r < 8; ++r) {
        const int M = r + (hi << 3);
        float zi = acc0[0][j][r] + bi;
        float zf = acc0[1][j][r] + bf;
        float zg = acc0[2][j][r] + bg;
        float zo = acc0[3][j][r] + bo;
        float cn = fast_sig(zf) * c0v[j][r] + fast_sig(zi) * fast_tanh(zg);
        float hn = fast_sig(zo) * fast_tanh(cn);
        c0v[j][r] = cn;
        _Float16 hh = (_Float16)hn;
        sXH0[M][F_IN + colb] = hh;   // next step's layer-0 recurrent input
        sXH1[M][colb]        = hh;   // this step's layer-1 input
      }
    }
    __syncthreads();

    // ================= layer 1 GEMM: z1 = [h0|h1] @ W1cat^T =================
    v8f acc1[4][NTW] = {};
    {
      const _Float16* arow = &sXH1[l15][0];
#pragma unroll 1
      for (int kb = 0; kb < KB1; ++kb) {
        union { v16h v; v8h h[2]; } a;
        a.h[0] = *(const v8h*)(arow + kb * 32 + hi * 8);
        a.h[1] = *(const v8h*)(arow + kb * 32 + hi * 8 + 16);
#pragma unroll
        for (int g = 0; g < 4; ++g) {
#pragma unroll
          for (int j = 0; j < NTW; ++j) {
            const int nt = g * 16 + (w << 1) + j;
            const _Float16* bp = pack1 + ((((size_t)nt * KB1 + kb) * 32 + lane) << 4);
            union { v16h v; v8h h[2]; } b;
            b.h[0] = *(const v8h*)bp;
            b.h[1] = *(const v8h*)(bp + 8);
            acc1[g][j] = __builtin_amdgcn_wmma_f32_16x16x32_f16(
                false, a.v, false, b.v, (short)0, acc1[g][j], false, false);
          }
        }
      }
    }
    __syncthreads();   // all waves done reading sXH1 before h1 overwrites

    // ================= layer 1 gates ========================================
#pragma unroll
    for (int j = 0; j < NTW; ++j) {
      const int colb = ((w << 1) + j) * 16 + l15;
      const float bi = sB1[colb];
      const float bf = sB1[256 + colb];
      const float bg = sB1[512 + colb];
      const float bo = sB1[768 + colb];
#pragma unroll
      for (int r = 0; r < 8; ++r) {
        const int M = r + (hi << 3);
        float zi = acc1[0][j][r] + bi;
        float zf = acc1[1][j][r] + bf;
        float zg = acc1[2][j][r] + bg;
        float zo = acc1[3][j][r] + bo;
        float cn = fast_sig(zf) * c1v[j][r] + fast_sig(zi) * fast_tanh(zg);
        float hn = fast_sig(zo) * fast_tanh(cn);
        c1v[j][r] = cn;
        sXH1[M][HID + colb] = (_Float16)hn;  // h1: fc input + next-step recurrent
      }
    }
    __syncthreads();

    // ====== fc head (parallel 16-thread/row dot) + window shift staging =====
    {
      float p = 0.0f;
#pragma unroll
      for (int k = 0; k < 16; ++k)
        p += (float)sXH1[fr][HID + fi * 16 + k] * sFC[fi * 16 + k];
      sRed[fr][fi] = p;

      // stage the autoregressive shift: thread fi owns xw slots j = fi*6..fi*6+5
      _Float16 stage[6];
#pragma unroll
      for (int q = 0; q < 6; ++q) {
        int jj = fi * 6 + q;
        stage[q] = (jj < F_IN - 1) ? sXH0[fr][jj + 1] : (_Float16)0.0f;
      }
      __syncthreads();

      float outv = fcb;
#pragma unroll
      for (int k = 0; k < 16; ++k) outv += sRed[fr][k];

      if (fi == 0)
        out[(size_t)(rowBase + fr) * STEPS + step] = outv;
#pragma unroll
      for (int q = 0; q < 6; ++q) {
        int jj = fi * 6 + q;
        sXH0[fr][jj] = (jj == F_IN - 1) ? (_Float16)outv : stage[q];
      }
    }
    __syncthreads();
  }
}

extern "C" void kernel_launch(void* const* d_in, const int* in_sizes, int n_in,
                              void* d_out, int out_size, void* d_ws, size_t ws_size,
                              hipStream_t stream) {
  (void)in_sizes; (void)n_in; (void)out_size; (void)ws_size;
  const float* x    = (const float*)d_in[0];
  const float* Wih0 = (const float*)d_in[1];
  const float* Whh0 = (const float*)d_in[2];
  const float* bih0 = (const float*)d_in[3];
  const float* bhh0 = (const float*)d_in[4];
  const float* Wih1 = (const float*)d_in[5];
  const float* Whh1 = (const float*)d_in[6];
  const float* bih1 = (const float*)d_in[7];
  const float* bhh1 = (const float*)d_in[8];
  const float* fcw  = (const float*)d_in[9];
  const float* fcb  = (const float*)d_in[10];
  float* out = (float*)d_out;

  _Float16* pack0 = (_Float16*)d_ws;
  _Float16* pack1 = pack0 + (size_t)PACK0_HALFS;

  const int totalPack = PACK0_HALFS + PACK1_HALFS;
  pack_weights_kernel<<<(totalPack + 255) / 256, 256, 0, stream>>>(
      Wih0, Whh0, Wih1, Whh1, pack0, pack1);

  lstm_persistent_kernel<<<BATCH / MT, 256, 0, stream>>>(
      pack0, pack1, x, bih0, bhh0, bih1, bhh1, fcw, fcb, out);
}
